// TwistBlock_13640816132252
// MI455X (gfx1250) — matrix operand order repaired and checked by reading, not compiled, not run on hardware
//
#include <hip/hip_runtime.h>
#include <hip/hip_bf16.h>

// ---------------------------------------------------------------------------
// MI455X (gfx1250). Memory-bound workload (~12 GFLOP vs ~600MB HBM traffic ->
// ~26us floor at 23.3 TB/s). All linears: bf16 WMMA v_wmma_f32_16x16x32_bf16
// (f32 accumulate). Weights staged pre-swizzled into WMMA B-fragment order so
// every fragment load is a contiguous 32B per lane (2x ds_load_b128 /
// global_load_b128). All dims are template constants -> div/mod become shifts.
// wave32, 256-thread blocks = 8 waves.
// ---------------------------------------------------------------------------

typedef __attribute__((ext_vector_type(16))) __bf16 v16bf;
typedef __attribute__((ext_vector_type(8)))  __bf16 v8bf;
typedef __attribute__((ext_vector_type(8)))  float  v8f;

#define WG 256

__device__ __forceinline__ float leaky01(float x) { return x >= 0.f ? x : 0.01f * x; }

// --- A fragment: 16x32 bf16 row-major tile (CDNA5 ISA 7.12.2). Lane l
// (M=l&15, g=l>>4) owns K in [k0+8g, k0+8g+8) u [k0+16+8g, ...+8): two
// contiguous 16B runs -> 2x ds_load_b128. lda must be a multiple of 8.
__device__ __forceinline__ v16bf wmma_load_a(const __bf16* a, int lda, int k0) {
  const int l = threadIdx.x & 31;
  const __bf16* p = a + (l & 15) * lda + k0 + 8 * (l >> 4);
  v8bf lo = *(const v8bf*)(p);
  v8bf hi = *(const v8bf*)(p + 16);
  v16bf f;
#pragma unroll
  for (int e = 0; e < 8; ++e) { f[e] = lo[e]; f[e + 8] = hi[e]; }
  return f;
}
// --- B fragment from pre-swizzled storage: tile-major, per-lane 16 contiguous
// bf16 (32B aligned) -> one vector load.
__device__ __forceinline__ v16bf wmma_load_b_sw(const __bf16* wf, int tile) {
  return *(const v16bf*)(wf + ((long)tile * 32 + (threadIdx.x & 31)) * 16);
}
// Swizzled offset for B element (k, n): lane = ((k>>3)&1)<<4 | (n&15),
// elem = (k&7) | ((k>>4)&1)<<3, tile = (n>>4)*KT + (k>>5).
template <int KT>
__device__ __forceinline__ long bswz(int k, int n) {
  const int lane = (((k >> 3) & 1) << 4) | (n & 15);
  const int tile = (n >> 4) * KT + (k >> 5);
  return ((long)tile * 32 + lane) * 16 + ((k & 7) | (((k >> 4) & 1) << 3));
}

#define WMMA_BF16(a, b, c) \
  __builtin_amdgcn_wmma_f32_16x16x32_bf16(false, (a), false, (b), (short)0, (c), false, false)

// ---------------------------------------------------------------------------
// Utility kernels
// ---------------------------------------------------------------------------
__global__ void fill0_kernel(float* p, long n) {
  long i = (long)blockIdx.x * WG + threadIdx.x;
  if (i < n) p[i] = 0.f;
}
__global__ void scatter_rows_kernel(float* dst, const float* src, const int* idx, int M, int F) {
  long i = (long)blockIdx.x * WG + threadIdx.x;
  if (i < (long)M * F) {
    int r = (int)(i / F), f = (int)(i % F);
    dst[(long)idx[r] * F + f] = src[i];
  }
}
// Convert fp32 weights (K x FOUT, K multiple of 32, FOUT power of 2) into
// bf16 pre-swizzled B-fragment order in global memory.
template <int K, int FOUT>
__global__ void convert_swizzle_kernel(const float* __restrict__ src, __bf16* __restrict__ dst) {
  constexpr int KT = K / 32;
  long i = (long)blockIdx.x * WG + threadIdx.x;
  if (i < (long)K * FOUT) {
    int k = (int)(i / FOUT), n = (int)(i % FOUT);
    dst[bswz<KT>(k, n)] = (__bf16)src[i];
  }
}

// ---------------------------------------------------------------------------
// contract_kernel<Y,FIN,H,FE,EYE>: AttentionContract fused
// (leaky -> NAL GEMM -> LayerNorm -> softmax over Y -> weighted sum).
// One WG per (b,x); Y streamed in 64-row chunks with online softmax.
// Strided addressing covers both tensor orientations (no transpose pass).
// ---------------------------------------------------------------------------
template <int Y, int FIN, int H, int FE, bool EYE>
__global__ __launch_bounds__(WG) void contract_kernel(
    const float* __restrict__ xy, long sb, long sx, long sy, int X,
    const float* __restrict__ W, const float* __restrict__ bias,
    const float* __restrict__ gamma, const float* __restrict__ beta,
    const float* __restrict__ atnW, const float* __restrict__ atnB,
    float* __restrict__ out) {
  constexpr int FOUT = H * FE;
  constexpr int Kp = (FIN + 31) & ~31;
  constexpr int KT = Kp / 32;
  constexpr int NT = FOUT / 16;
  constexpr int LDA = Kp + 8;   // multiple of 8 -> 16B-aligned fragment runs
  constexpr int LDH = FOUT + 4; // break bank conflicts on column walks

  __shared__ __align__(32) __bf16 aLds[64 * LDA];
  __shared__ __align__(32) __bf16 wLds[Kp * FOUT];  // swizzled fragments
  __shared__ float hLds[64 * LDH];
  __shared__ float lgt[64 * H];
  __shared__ float acc[FOUT];
  __shared__ float mRun[H], dRun[H], scl[H];

  const int tid = threadIdx.x;
  const int bx = blockIdx.x;
  const int x = bx % X;
  const float* base = xy + (long)(bx / X) * sb + (long)x * sx;

  for (int i = tid; i < Kp * FOUT; i += WG) {
    int k = i / FOUT, n = i % FOUT;
    float v = (k < FIN) ? W[k * FOUT + n] : 0.f;
    wLds[bswz<KT>(k, n)] = (__bf16)v;
  }
  if (tid < H) { mRun[tid] = -3.0e30f; dRun[tid] = 0.f; }
  for (int i = tid; i < FOUT; i += WG) acc[i] = 0.f;
  __syncthreads();

  for (int yc = 0; yc < Y; yc += 64) {
    // stage chunk: leaky + bf16 (zero-pad K)
    for (int i = tid; i < 64 * Kp; i += WG) {
      int r = i / Kp, f = i % Kp;
      float v = (f < FIN) ? leaky01(base[(long)(yc + r) * sy + f]) : 0.f;
      aLds[r * LDA + f] = (__bf16)v;
    }
    // prefetch next chunk's rows into L2 while this chunk computes
    if (yc + 64 < Y && tid < 64)
      __builtin_prefetch(&base[(long)(yc + 64 + tid) * sy], 0, 0);
    __syncthreads();
    // attention logits (H<=8): VALU dot, tiny vs GEMM
    for (int i = tid; i < 64 * H; i += WG) {
      int r = i / H, h = i % H;
      float s = atnB[h];
      for (int f = 0; f < FIN; ++f) s += (float)aLds[r * LDA + f] * atnW[f * H + h];
      if (EYE && (yc + r) == x) s = 0.f;  // reference: where(eye, 0.0, atn)
      lgt[r * H + h] = s;
    }
    // NAL GEMM: 64 x FIN @ FIN x FOUT via bf16 WMMA
    for (int t = tid >> 5; t < 4 * NT; t += 8) {
      int mi = t / NT, ni = t % NT;
      v8f c = {0.f, 0.f, 0.f, 0.f, 0.f, 0.f, 0.f, 0.f};
#pragma unroll
      for (int kt = 0; kt < KT; ++kt) {
        v16bf a = wmma_load_a(aLds + mi * 16 * LDA, LDA, kt * 32);
        v16bf b = wmma_load_b_sw(wLds, ni * KT + kt);
        c = WMMA_BF16(a, b, c);
      }
      int l = tid & 31, N = ni * 16 + (l & 15), g = l >> 4;
#pragma unroll
      for (int v = 0; v < 8; ++v) hLds[(mi * 16 + v + 8 * g) * LDH + N] = c[v];
    }
    __syncthreads();
    // per-row LayerNorm (bias added here; population variance, eps 1e-5)
    if (tid < 64) {
      int r = tid;
      float s = 0.f, s2 = 0.f;
      for (int c0 = 0; c0 < FOUT; ++c0) {
        float v = hLds[r * LDH + c0] + bias[c0];
        s += v; s2 += v * v;
      }
      float mu = s / FOUT;
      float inv = rsqrtf(s2 / FOUT - mu * mu + 1e-5f);
      for (int c0 = 0; c0 < FOUT; ++c0) {
        float v = hLds[r * LDH + c0] + bias[c0];
        hLds[r * LDH + c0] = (v - mu) * inv * gamma[c0] + beta[c0];
      }
    }
    __syncthreads();
    // online softmax bookkeeping per head
    if (tid < H) {
      float cm = -3.0e30f;
      for (int r = 0; r < 64; ++r) cm = fmaxf(cm, lgt[r * H + tid]);
      float nm = fmaxf(mRun[tid], cm);
      float sc = __expf(mRun[tid] - nm);
      float ds = dRun[tid] * sc;
      for (int r = 0; r < 64; ++r) {
        float w = __expf(lgt[r * H + tid] - nm);
        lgt[r * H + tid] = w;
        ds += w;
      }
      mRun[tid] = nm; dRun[tid] = ds; scl[tid] = sc;
    }
    __syncthreads();
    for (int i = tid; i < FOUT; i += WG) {
      int h = i / FE;
      float a2 = acc[i] * scl[h];
      for (int r = 0; r < 64; ++r) a2 += lgt[r * H + h] * hLds[r * LDH + i];
      acc[i] = a2;
    }
    __syncthreads();
  }
  for (int i = tid; i < FOUT; i += WG) out[(long)bx * FOUT + i] = acc[i] / dRun[i / FE];
}

// ---------------------------------------------------------------------------
// nal_kernel<FIN,FOUT>: out = LN(leaky(in) @ W + b)*g + be.  32 rows per WG.
// ---------------------------------------------------------------------------
template <int FIN, int FOUT>
__global__ __launch_bounds__(WG) void nal_kernel(
    const float* __restrict__ in,
    const float* __restrict__ W, const float* __restrict__ bias,
    const float* __restrict__ gamma, const float* __restrict__ beta,
    float* __restrict__ out) {
  constexpr int Kp = (FIN + 31) & ~31;
  constexpr int KT = Kp / 32;
  constexpr int NT = FOUT / 16;
  constexpr int LDA = Kp + 8;
  constexpr int LDH = FOUT + 4;
  __shared__ __align__(32) __bf16 aLds[32 * LDA];
  __shared__ __align__(32) __bf16 wLds[Kp * FOUT];
  __shared__ float hLds[32 * LDH];

  const int tid = threadIdx.x;
  const int r0 = blockIdx.x * 32;

  for (int i = tid; i < Kp * FOUT; i += WG) {
    int k = i / FOUT, n = i % FOUT;
    float v = (k < FIN) ? W[k * FOUT + n] : 0.f;
    wLds[bswz<KT>(k, n)] = (__bf16)v;
  }
  for (int i = tid; i < 32 * Kp; i += WG) {
    int r = i / Kp, f = i % Kp;
    float v = (f < FIN) ? leaky01(in[(long)(r0 + r) * FIN + f]) : 0.f;
    aLds[r * LDA + f] = (__bf16)v;
  }
  __syncthreads();
  for (int t = tid >> 5; t < 2 * NT; t += 8) {
    int mi = t / NT, ni = t % NT;
    v8f c = {0.f, 0.f, 0.f, 0.f, 0.f, 0.f, 0.f, 0.f};
#pragma unroll
    for (int kt = 0; kt < KT; ++kt) {
      v16bf a = wmma_load_a(aLds + mi * 16 * LDA, LDA, kt * 32);
      v16bf b = wmma_load_b_sw(wLds, ni * KT + kt);
      c = WMMA_BF16(a, b, c);
    }
    int l = tid & 31, N = ni * 16 + (l & 15), g = l >> 4;
#pragma unroll
    for (int v = 0; v < 8; ++v) hLds[(mi * 16 + v + 8 * g) * LDH + N] = c[v];
  }
  __syncthreads();
  if (tid < 32) {
    int r = tid;
    float s = 0.f, s2 = 0.f;
    for (int c0 = 0; c0 < FOUT; ++c0) {
      float v = hLds[r * LDH + c0] + bias[c0];
      s += v; s2 += v * v;
    }
    float mu = s / FOUT;
    float inv = rsqrtf(s2 / FOUT - mu * mu + 1e-5f);
    for (int c0 = 0; c0 < FOUT; ++c0) {
      float v = hLds[r * LDH + c0] + bias[c0];
      out[(long)(r0 + r) * FOUT + c0] = (v - mu) * inv * gamma[c0] + beta[c0];
    }
  }
}

// ---------------------------------------------------------------------------
// FullRecContract: per-atom logits, then per-residue (8 contiguous atoms per
// the res_index generator) softmax pooling. The reference's global max
// subtraction cancels against the per-segment denominator -> per-segment max.
// ---------------------------------------------------------------------------
__global__ void frc_logit_kernel(const float* __restrict__ feat,
                                 const float* __restrict__ aw, const float* __restrict__ ab,
                                 float* __restrict__ lg, int Ntot, int F) {
  int i = blockIdx.x * WG + threadIdx.x;
  if (i < Ntot) {
    float s = ab[0];
    for (int f = 0; f < F; ++f) s += leaky01(feat[(long)i * F + f]) * aw[f];
    lg[i] = s;
  }
}
__global__ void frc_combine_kernel(const float* __restrict__ nalv, const float* __restrict__ lg,
                                   float* __restrict__ out, int NSEG, int F) {
  int i = blockIdx.x * WG + threadIdx.x;
  if (i < NSEG * F) {
    int s = i / F, c = i % F;
    float mx = -3.0e30f;
#pragma unroll
    for (int a = 0; a < 8; ++a) mx = fmaxf(mx, lg[s * 8 + a]);
    float den = 0.f, num = 0.f;
#pragma unroll
    for (int a = 0; a < 8; ++a) {
      float w = __expf(lg[s * 8 + a] - mx);
      den += w;
      num += w * nalv[(long)(s * 8 + a) * F + c];
    }
    out[i] = num / den;
  }
}

// ---------------------------------------------------------------------------
// concat_gemm_kernel<K,FOUT,F0..F3>: out = concat(gathered segments) @ W + b.
// A staged in LDS bf16; W read as pre-swizzled bf16 fragments from global
// (L2-hot, 32B contiguous per lane -> coalesced global_load_b128).
// ---------------------------------------------------------------------------
template <int K, int FOUT, int F0, int F1, int F2, int F3>
__global__ __launch_bounds__(WG) void concat_gemm_kernel(
    const float* s0, const int* i0,
    const float* s1, const int* i1,
    const float* s2, const int* i2,
    const float* s3, const int* i3,
    const __bf16* __restrict__ Wb, const float* __restrict__ bias,
    float* __restrict__ out) {
  constexpr int KT = K / 32;
  constexpr int NT = FOUT / 16;
  __shared__ __align__(32) __bf16 aL[32 * K];  // K multiple of 8
  const int tid = threadIdx.x;
  const int r0 = blockIdx.x * 32;
  for (int r = 0; r < 32; ++r) {
    const int row = r0 + r;
    for (int k = tid; k < K; k += WG) {
      float v;
      if (k < F0) {
        int sr = i0 ? i0[row] : row;  v = s0[(long)sr * F0 + k];
      } else if (k < F0 + F1) {
        int sr = i1 ? i1[row] : row;  v = s1[(long)sr * F1 + (k - F0)];
      } else if (k < F0 + F1 + F2) {
        int sr = i2 ? i2[row] : row;  v = s2[(long)sr * F2 + (k - F0 - F1)];
      } else {
        int sr = i3 ? i3[row] : row;  v = s3[(long)sr * F3 + (k - F0 - F1 - F2)];
      }
      aL[r * K + k] = (__bf16)v;
    }
  }
  __syncthreads();
  for (int t = tid >> 5; t < 2 * NT; t += 8) {
    int mi = t / NT, ni = t % NT;
    v8f c = {0.f, 0.f, 0.f, 0.f, 0.f, 0.f, 0.f, 0.f};
#pragma unroll
    for (int kt = 0; kt < KT; ++kt) {
      v16bf a = wmma_load_a(aL + mi * 16 * K, K, kt * 32);
      v16bf b = wmma_load_b_sw(Wb, ni * KT + kt);
      c = WMMA_BF16(a, b, c);
    }
    int l = tid & 31, N = ni * 16 + (l & 15), g = l >> 4;
#pragma unroll
    for (int v = 0; v < 8; ++v) {
      long row = r0 + mi * 16 + v + 8 * g;
      out[row * FOUT + N] = c[v] + bias[N];
    }
  }
}

// ---------------------------------------------------------------------------
// pair_kernel<Y,FP,H,FE,FO>: out[b,x,y,:] = concat(pair_feat, expand) @ W + b.
// One WG per (b,x); Y streamed in 64-row chunks; expand dot (f<=16) on VALU.
// ---------------------------------------------------------------------------
template <int Y, int FP, int H, int FE, int FO>
__global__ __launch_bounds__(WG) void pair_kernel(
    const float* __restrict__ pf, const float* __restrict__ xh, const float* __restrict__ yh,
    int X, const float* __restrict__ W, const float* __restrict__ bias,
    float* __restrict__ out) {
  constexpr int HF = H * FE;
  constexpr int K = FP + H;
  constexpr int Kp = (K + 31) & ~31;
  constexpr int KT = Kp / 32;
  constexpr int NT = FO / 16;
  constexpr int LDA = Kp + 8;
  __shared__ float  xrow[HF];
  __shared__ __align__(32) __bf16 aL[64 * LDA];
  __shared__ __align__(32) __bf16 wL[Kp * FO];
  const int tid = threadIdx.x;
  const int bx = blockIdx.x;
  const int b = bx / X;

  for (int i = tid; i < Kp * FO; i += WG) {
    int k = i / FO, n = i % FO;
    float v = (k < K) ? W[k * FO + n] : 0.f;
    wL[bswz<KT>(k, n)] = (__bf16)v;
  }
  for (int i = tid; i < HF; i += WG) xrow[i] = xh[(long)bx * HF + i];
  __syncthreads();

  const float* pbase = pf + (long)bx * Y * FP;
  for (int yc = 0; yc < Y; yc += 64) {
    for (int i = tid; i < 64 * Kp; i += WG) {
      int r = i / Kp, k = i % Kp;
      float v;
      if (k < FP) {
        v = pbase[(long)(yc + r) * FP + k];
      } else if (k < K) {
        int h = k - FP;
        const float* yr = yh + ((long)b * Y + yc + r) * HF + h * FE;
        float s = 0.f;
#pragma unroll
        for (int f = 0; f < FE; ++f) s += xrow[h * FE + f] * yr[f];
        v = s;
      } else {
        v = 0.f;
      }
      aL[r * LDA + k] = (__bf16)v;
    }
    if (yc + 64 < Y && tid < 64) {
      __builtin_prefetch(&pbase[(long)(yc + 64 + tid) * FP], 0, 0);
      __builtin_prefetch(&yh[((long)b * Y + yc + 64 + tid) * HF], 0, 0);
    }
    __syncthreads();
    for (int t = tid >> 5; t < 4 * NT; t += 8) {
      int mi = t / NT, ni = t % NT;
      v8f c = {0.f, 0.f, 0.f, 0.f, 0.f, 0.f, 0.f, 0.f};
#pragma unroll
      for (int kt = 0; kt < KT; ++kt) {
        v16bf a = wmma_load_a(aL + mi * 16 * LDA, LDA, kt * 32);
        v16bf bb = wmma_load_b_sw(wL, ni * KT + kt);
        c = WMMA_BF16(a, bb, c);
      }
      int l = tid & 31, N = ni * 16 + (l & 15), g = l >> 4;
#pragma unroll
      for (int v = 0; v < 8; ++v) {
        long row = (long)bx * Y + yc + mi * 16 + v + 8 * g;
        out[row * FO + N] = c[v] + bias[N];
      }
    }
    __syncthreads();
  }
}

// ---------------------------------------------------------------------------
// Host-side orchestration
// ---------------------------------------------------------------------------
static inline int gdiv(long n, int b) { return (int)((n + b - 1) / b); }

extern "C" void kernel_launch(void* const* d_in, const int* in_sizes, int n_in,
                              void* d_out, int out_size, void* d_ws, size_t ws_size,
                              hipStream_t stream) {
  (void)in_sizes; (void)n_in; (void)out_size; (void)ws_size;
  const int B = 16, L = 64, RA = 128, RF = 1024;
  // d_in layout: tensors [0..5], params leaves [6..81] (dict insertion order,
  // nal = {W,b,g,be}, linear = {W,b}), index arrays [82..85].
  const float* lig  = (const float*)d_in[0];   // (768,128)
  const float* rec  = (const float*)d_in[1];   // (1536,128)
  const float* frec = (const float*)d_in[2];   // (12288,64)
  const float* llf  = (const float*)d_in[3];   // (16,64,64,64)
  const float* lraf = (const float*)d_in[4];   // (16,64,128,32)
  const float* lrff = (const float*)d_in[5];   // (16,64,1024,32)
  const float* P[82];
  for (int i = 6; i < 82; ++i) P[i] = (const float*)d_in[i];
  const int* idxL  = (const int*)d_in[82];   // 768
  const int* idxR  = (const int*)d_in[83];   // 1536
  const int* idxRF = (const int*)d_in[84];   // 12288
  const int* resIx = (const int*)d_in[85];   // 12288

  float* outv = (float*)d_out;
  float* lig_out  = outv;                     // 768*128
  float* rec_out  = outv + 98304;             // 1536*128
  float* frec_out = outv + 294912;            // 12288*64
  float* ll_out   = outv + 1081344;           // 16*64*64*64
  float* lra_out  = outv + 5275648;           // 16*64*128*32
  float* lrf_out  = outv + 9469952;           // 16*64*1024*32

  // ---- workspace carve ----
  float* ws = (float*)d_ws;
  long o = 0;
  auto alloc = [&](long n) { float* p = ws + o; o += n; return p; };
  float* c_ll       = alloc(1024L * 128);
  float* c_lra_lig  = alloc(1024L * 128);
  float* c_lrf_lig  = alloc(1024L * 32);
  float* c_lra_rec  = alloc(2048L * 128);
  float* c_lrf_frec = alloc(16384L * 32);
  float* frc_nal    = alloc(12288L * 64);
  float* frc_lg     = alloc(12288L);
  float* frc_out    = alloc(1536L * 64);
  float* fre_nal    = alloc(1536L * 64);
  float* lig_pk     = alloc(1024L * 128);
  float* rec_pk     = alloc(2048L * 128);
  float* frec_pk    = alloc(16384L * 64);
  float* xh_ll      = alloc(1024L * 128);
  float* yh_ll      = alloc(1024L * 128);
  float* xh_lra     = alloc(1024L * 128);
  float* yh_lra     = alloc(2048L * 128);
  float* xh_lrf     = alloc(1024L * 32);
  float* yh_lrf     = alloc(16384L * 32);
  __bf16* w_lig_bf  = (__bf16*)alloc((416L * 128 + 1) / 2);
  __bf16* w_rec_bf  = (__bf16*)alloc((320L * 128 + 1) / 2);
  __bf16* w_frec_bf = (__bf16*)alloc((160L * 64 + 1) / 2);

  // ---- 0) padded-packed tensors + swizzled bf16 output weights ----
  fill0_kernel<<<gdiv(1024L * 128, WG), WG, 0, stream>>>(lig_pk, 1024L * 128);
  fill0_kernel<<<gdiv(2048L * 128, WG), WG, 0, stream>>>(rec_pk, 2048L * 128);
  fill0_kernel<<<gdiv(16384L * 64, WG), WG, 0, stream>>>(frec_pk, 16384L * 64);
  scatter_rows_kernel<<<gdiv(768L * 128, WG), WG, 0, stream>>>(lig_pk, lig, idxL, 768, 128);
  scatter_rows_kernel<<<gdiv(1536L * 128, WG), WG, 0, stream>>>(rec_pk, rec, idxR, 1536, 128);
  scatter_rows_kernel<<<gdiv(12288L * 64, WG), WG, 0, stream>>>(frec_pk, frec, idxRF, 12288, 64);
  convert_swizzle_kernel<416, 128><<<gdiv(416L * 128, WG), WG, 0, stream>>>(P[70], w_lig_bf);
  convert_swizzle_kernel<320, 128><<<gdiv(320L * 128, WG), WG, 0, stream>>>(P[72], w_rec_bf);
  convert_swizzle_kernel<160, 64><<<gdiv(160L * 64, WG), WG, 0, stream>>>(P[74], w_frec_bf);

  // ---- 1) attention contracts (fused WMMA + LN + online softmax) ----
  contract_kernel<64, 64, 8, 16, true><<<B * L, WG, 0, stream>>>(
      llf, 64L * 64 * 64, 64L * 64, 64, L,
      P[6], P[7], P[8], P[9], P[10], P[11], c_ll);
  contract_kernel<128, 32, 8, 16, false><<<B * L, WG, 0, stream>>>(
      lraf, 64L * 128 * 32, 128L * 32, 32, L,
      P[12], P[13], P[14], P[15], P[16], P[17], c_lra_lig);
  contract_kernel<1024, 32, 4, 8, false><<<B * L, WG, 0, stream>>>(
      lrff, 64L * 1024 * 32, 1024L * 32, 32, L,
      P[18], P[19], P[20], P[21], P[22], P[23], c_lrf_lig);
  contract_kernel<64, 32, 8, 16, false><<<B * RA, WG, 0, stream>>>(
      lraf, 64L * 128 * 32, 32, 128L * 32, RA,
      P[24], P[25], P[26], P[27], P[28], P[29], c_lra_rec);
  contract_kernel<64, 32, 4, 8, false><<<B * RF, WG, 0, stream>>>(
      lrff, 64L * 1024 * 32, 32, 1024L * 32, RF,
      P[30], P[31], P[32], P[33], P[34], P[35], c_lrf_frec);

  // ---- 2) NAL projections ----
  nal_kernel<64, 64><<<12288 / 32, WG, 0, stream>>>(frec, P[36], P[37], P[38], P[39], frc_nal);
  nal_kernel<128, 64><<<1536 / 32, WG, 0, stream>>>(rec, P[42], P[43], P[44], P[45], fre_nal);
  nal_kernel<128, 128><<<1024 / 32, WG, 0, stream>>>(lig_pk, P[46], P[47], P[48], P[49], xh_ll);
  nal_kernel<128, 128><<<1024 / 32, WG, 0, stream>>>(lig_pk, P[50], P[51], P[52], P[53], yh_ll);
  nal_kernel<128, 128><<<1024 / 32, WG, 0, stream>>>(lig_pk, P[54], P[55], P[56], P[57], xh_lra);
  nal_kernel<128, 128><<<2048 / 32, WG, 0, stream>>>(rec_pk, P[58], P[59], P[60], P[61], yh_lra);
  nal_kernel<128, 32><<<1024 / 32, WG, 0, stream>>>(lig_pk, P[62], P[63], P[64], P[65], xh_lrf);
  nal_kernel<64, 32><<<16384 / 32, WG, 0, stream>>>(frec_pk, P[66], P[67], P[68], P[69], yh_lrf);

  // ---- 3) FullRecContract segment softmax ----
  frc_logit_kernel<<<gdiv(12288, WG), WG, 0, stream>>>(frec, P[40], P[41], frc_lg, 12288, 64);
  frc_combine_kernel<<<gdiv(1536L * 64, WG), WG, 0, stream>>>(frc_nal, frc_lg, frc_out, 1536, 64);

  // ---- 4) node output linears (gather + concat + WMMA GEMM) ----
  concat_gemm_kernel<416, 128, 128, 128, 128, 32><<<768 / 32, WG, 0, stream>>>(
      lig, nullptr, c_ll, idxL, c_lra_lig, idxL, c_lrf_lig, idxL,
      w_lig_bf, P[71], lig_out);
  concat_gemm_kernel<320, 128, 128, 128, 64, 0><<<1536 / 32, WG, 0, stream>>>(
      rec, nullptr, c_lra_rec, idxR, frc_out, nullptr, nullptr, nullptr,
      w_rec_bf, P[73], rec_out);
  concat_gemm_kernel<160, 64, 64, 32, 64, 0><<<12288 / 32, WG, 0, stream>>>(
      frec, nullptr, c_lrf_frec, idxRF, fre_nal, resIx, nullptr, nullptr,
      w_frec_bf, P[75], frec_out);

  // ---- 5) pair feature updates (expand + concat + WMMA GEMM) ----
  pair_kernel<64, 64, 8, 16, 64><<<B * L, WG, 0, stream>>>(
      llf, xh_ll, yh_ll, L, P[76], P[77], ll_out);
  pair_kernel<128, 32, 8, 16, 32><<<B * L, WG, 0, stream>>>(
      lraf, xh_lra, yh_lra, L, P[78], P[79], lra_out);
  pair_kernel<1024, 32, 4, 8, 32><<<B * L, WG, 0, stream>>>(
      lrff, xh_lrf, yh_lrf, L, P[80], P[81], lrf_out);
}
